// QATGPT2Attention_28286654611542
// MI455X (gfx1250) — matrix operand order repaired and checked by reading, not compiled, not run on hardware
//
#include <hip/hip_runtime.h>

typedef __bf16 bf16;
typedef __attribute__((ext_vector_type(16))) __bf16 v16bf;
typedef __attribute__((ext_vector_type(8)))  __bf16 v8bf;
typedef __attribute__((ext_vector_type(8)))  float  v8f;

#define WMMA_BF16(a,b,c) \
  __builtin_amdgcn_wmma_f32_16x16x32_bf16(false,(a),false,(b),(short)0,(c),false,false)

static __device__ __forceinline__ v16bf cat8(v8bf lo, v8bf hi) {
  return __builtin_shufflevector(lo, hi, 0,1,2,3,4,5,6,7,8,9,10,11,12,13,14,15);
}

// ---------------------------------------------------------------- constants
// B=2, T=2048, C=1024, H=16, D=64, M = B*T = 4096
#define M_TOK 4096
#define T_SEQ 2048
#define C_DIM 1024
#define N_QKV 3072
#define H_HEADS 16
#define D_HEAD 64

// ---------------------------------------------------------------- pre-kernels
__global__ void cvt_f32_bf16(const float* __restrict__ in, bf16* __restrict__ out, int n) {
  int i = blockIdx.x * blockDim.x + threadIdx.x;
  if (i < n) out[i] = (bf16)in[i];
}

// in[K][N] (fp32 row-major) -> out[N][K] (bf16 row-major)
__global__ void transpose_f32_bf16(const float* __restrict__ in, bf16* __restrict__ out,
                                   int K, int N) {
  int i = blockIdx.x * blockDim.x + threadIdx.x;
  if (i < K * N) {
    int k = i / N, n = i - k * N;
    out[n * K + k] = (bf16)in[i];
  }
}

// ---------------------------------------------------------------- QKV GEMM + fake-quant
// Wave tile 32(M) x 64(N): 2 A-fragments reused against 4 B-fragments
// => 8 WMMA per 12 b128 loads per k-step.
// q  -> qb [B,H,T,D] bf16
// k  -> fake_quant -> kb [B,H,T,D] bf16
// v  -> fake_quant -> vt [B,H,D,T] bf16 (transposed for PV WMMA B-fragments)
__global__ void __launch_bounds__(256) qkv_wmma(
    const bf16* __restrict__ X, const bf16* __restrict__ Wt,
    const float* __restrict__ bias,
    const float* __restrict__ sc_p, const float* __restrict__ zp_p,
    bf16* __restrict__ qb, bf16* __restrict__ kb, bf16* __restrict__ vt) {
  const int lane = threadIdx.x & 31, wave = threadIdx.x >> 5;
  const int lg = lane >> 4, lm = lane & 15;
  const int gw = blockIdx.x * 8 + wave;          // global wave id
  const int NT = N_QKV / 64;                     // 48 column tiles of width 64
  const int mt = gw / NT, nt = gw - mt * NT;     // 128 x 48
  const int m0 = mt * 32, n0 = nt * 64;

  const bf16* arow0 = X + (m0 + lm) * C_DIM;
  const bf16* arow1 = X + (m0 + 16 + lm) * C_DIM;
  v8f c00 = {}, c01 = {}, c02 = {}, c03 = {};
  v8f c10 = {}, c11 = {}, c12 = {}, c13 = {};
#pragma unroll 2
  for (int k = 0; k < C_DIM; k += 32) {
    v16bf a0 = cat8(*(const v8bf*)(arow0 + k + 16 * lg),
                    *(const v8bf*)(arow0 + k + 16 * lg + 8));
    v16bf a1 = cat8(*(const v8bf*)(arow1 + k + 16 * lg),
                    *(const v8bf*)(arow1 + k + 16 * lg + 8));
    const bf16* bb = Wt + (n0 + lm) * C_DIM + k + 16 * lg;
#if defined(__has_builtin)
#if __has_builtin(__builtin_prefetch)
    __builtin_prefetch(bb + 32, 0, 1);           // next k-step B rows -> global_prefetch
#endif
#endif
    v16bf b0 = *(const v16bf*)(bb);
    v16bf b1 = *(const v16bf*)(bb + 16 * C_DIM);
    v16bf b2 = *(const v16bf*)(bb + 32 * C_DIM);
    v16bf b3 = *(const v16bf*)(bb + 48 * C_DIM);
    c00 = WMMA_BF16(a0, b0, c00);  c01 = WMMA_BF16(a0, b1, c01);
    c02 = WMMA_BF16(a0, b2, c02);  c03 = WMMA_BF16(a0, b3, c03);
    c10 = WMMA_BF16(a1, b0, c10);  c11 = WMMA_BF16(a1, b1, c11);
    c12 = WMMA_BF16(a1, b2, c12);  c13 = WMMA_BF16(a1, b3, c13);
  }

  const float scale = sc_p[0], zp = zp_p[0];
  const int section = n0 >> 10;                 // 0=q 1=k 2=v
  const int h = (n0 & 1023) >> 6;
  v8f cc[2][4] = {{c00, c01, c02, c03}, {c10, c11, c12, c13}};
#pragma unroll
  for (int i = 0; i < 2; ++i) {
#pragma unroll
    for (int dt = 0; dt < 4; ++dt) {
      const int n = n0 + dt * 16 + lm;
      const float bv = bias[n];
      const int d = dt * 16 + lm;               // n0 is 64-aligned -> d within head
#pragma unroll
      for (int r = 0; r < 8; ++r) {
        const int m = m0 + 16 * i + r + 8 * lg;
        const int b = m >> 11, t = m & (T_SEQ - 1);
        float val = cc[i][dt][r] + bv;
        if (section == 0) {
          qb[(((b * H_HEADS + h) * T_SEQ) + t) * D_HEAD + d] = (bf16)val;
        } else {
          float qv = rintf(val / scale + zp);   // round-half-even like jnp.round
          qv = fminf(fmaxf(qv, 0.f), 255.f);
          float fq = (qv - zp) * scale;
          if (section == 1)
            kb[(((b * H_HEADS + h) * T_SEQ) + t) * D_HEAD + d] = (bf16)fq;
          else
            vt[(((b * H_HEADS + h) * D_HEAD) + d) * T_SEQ + t] = (bf16)fq;
        }
      }
    }
  }
}

// ---------------------------------------------------------------- flash attention
// One wave = 16 query rows. Online softmax, P staged through per-wave LDS to
// convert C-layout -> A-fragment layout. Causal early exit (no block barriers).
__global__ void __launch_bounds__(256) attn_wmma(
    const bf16* __restrict__ qb, const bf16* __restrict__ kb,
    const bf16* __restrict__ vt, bf16* __restrict__ att) {
  __shared__ __align__(16) bf16 Pb[8][16][32];
  const int lane = threadIdx.x & 31, wave = threadIdx.x >> 5;
  const int lg = lane >> 4, lm = lane & 15;
  const int bh = blockIdx.x >> 4;               // B*H = 32
  const int qblk = blockIdx.x & 15;             // T / 128 = 16
  const int b = bh >> 4, h = bh & 15;
  const int q0 = qblk * 128 + wave * 16;

  const bf16* qbase = qb + (size_t)bh * T_SEQ * D_HEAD;
  const bf16* kbase = kb + (size_t)bh * T_SEQ * D_HEAD;
  const bf16* vbase = vt + (size_t)bh * D_HEAD * T_SEQ;

  // Q A-fragments for d chunks [0,32) and [32,64): reused every iteration
  const bf16* qrow = qbase + (q0 + lm) * D_HEAD;
  v16bf qa0 = cat8(*(const v8bf*)(qrow + 0 + 16 * lg),  *(const v8bf*)(qrow + 8 + 16 * lg));
  v16bf qa1 = cat8(*(const v8bf*)(qrow + 32 + 16 * lg), *(const v8bf*)(qrow + 40 + 16 * lg));

  v8f o[4] = {};
  float mrow[8], lrow[8];
#pragma unroll
  for (int r = 0; r < 8; ++r) { mrow[r] = -1e30f; lrow[r] = 0.f; }

  const int qmax = q0 + 15;
  const float sm_scale = 0.125f;                // 1/sqrt(64)

  for (int kt = 0; kt <= qmax; kt += 32) {
    // S tiles: keys [kt,kt+16) and [kt+16,kt+32); contraction over d
    const bf16* k0 = kbase + (kt + lm) * D_HEAD + 16 * lg;
    const bf16* k1 = k0 + 16 * D_HEAD;
    v8f s0 = {}, s1 = {};
    s0 = WMMA_BF16(qa0, *(const v16bf*)(k0),      s0);
    s0 = WMMA_BF16(qa1, *(const v16bf*)(k0 + 32), s0);
    s1 = WMMA_BF16(qa0, *(const v16bf*)(k1),      s1);
    s1 = WMMA_BF16(qa1, *(const v16bf*)(k1 + 32), s1);

#pragma unroll
    for (int r = 0; r < 8; ++r) {
      const int mq = q0 + r + 8 * lg;                // global query index
      float e0 = s0[r] * sm_scale;
      float e1 = s1[r] * sm_scale;
      if (kt + lm > mq)      e0 = -1e30f;            // causal mask
      if (kt + 16 + lm > mq) e1 = -1e30f;
      float mx = fmaxf(e0, e1);
#pragma unroll
      for (int off = 8; off >= 1; off >>= 1)         // reduce over 16-lane half-group
        mx = fmaxf(mx, __shfl_xor(mx, off, 32));
      const float mn = fmaxf(mrow[r], mx);
      const float alpha = __expf(mrow[r] - mn);
      const float p0 = __expf(e0 - mn);
      const float p1 = __expf(e1 - mn);
      float sum = p0 + p1;
#pragma unroll
      for (int off = 8; off >= 1; off >>= 1)
        sum += __shfl_xor(sum, off, 32);
      lrow[r] = lrow[r] * alpha + sum;
      mrow[r] = mn;
      o[0][r] *= alpha; o[1][r] *= alpha; o[2][r] *= alpha; o[3][r] *= alpha;
      Pb[wave][r + 8 * lg][lm]      = (bf16)p0;      // C-layout -> LDS
      Pb[wave][r + 8 * lg][16 + lm] = (bf16)p1;
    }

    // P A-fragment (same wave wrote it; DS ops are in-order per wave)
    v16bf pa = cat8(*(const v8bf*)&Pb[wave][lm][16 * lg],
                    *(const v8bf*)&Pb[wave][lm][16 * lg + 8]);
    // V B-fragments: contraction over key, N = d; vt is [d][t] row-major
#pragma unroll
    for (int dt = 0; dt < 4; ++dt) {
      v16bf vb = *(const v16bf*)(vbase + (dt * 16 + lm) * T_SEQ + kt + 16 * lg);
      o[dt] = WMMA_BF16(pa, vb, o[dt]);
    }
  }

  // normalize and store to att [B,T,C] bf16
#pragma unroll
  for (int dt = 0; dt < 4; ++dt) {
#pragma unroll
    for (int r = 0; r < 8; ++r) {
      const int tok = b * T_SEQ + q0 + r + 8 * lg;
      const int col = h * D_HEAD + dt * 16 + lm;
      att[(size_t)tok * C_DIM + col] = (bf16)(o[dt][r] / lrow[r]);
    }
  }
}

// ---------------------------------------------------------------- output projection
// Wave tile 32(M) x 64(N). A [4096,1024] bf16, Wt [1024,1024] bf16 (W_proj^T),
// bias fp32 -> out fp32 [B,T,C]
__global__ void __launch_bounds__(256) proj_wmma(
    const bf16* __restrict__ A, const bf16* __restrict__ Wt,
    const float* __restrict__ bias, float* __restrict__ out) {
  const int lane = threadIdx.x & 31, wave = threadIdx.x >> 5;
  const int lg = lane >> 4, lm = lane & 15;
  const int gw = blockIdx.x * 8 + wave;
  const int NT = C_DIM / 64;                    // 16
  const int mt = gw / NT, nt = gw - mt * NT;    // 128 x 16
  const int m0 = mt * 32, n0 = nt * 64;

  const bf16* arow0 = A + (m0 + lm) * C_DIM;
  const bf16* arow1 = A + (m0 + 16 + lm) * C_DIM;
  v8f c00 = {}, c01 = {}, c02 = {}, c03 = {};
  v8f c10 = {}, c11 = {}, c12 = {}, c13 = {};
#pragma unroll 2
  for (int k = 0; k < C_DIM; k += 32) {
    v16bf a0 = cat8(*(const v8bf*)(arow0 + k + 16 * lg),
                    *(const v8bf*)(arow0 + k + 16 * lg + 8));
    v16bf a1 = cat8(*(const v8bf*)(arow1 + k + 16 * lg),
                    *(const v8bf*)(arow1 + k + 16 * lg + 8));
    const bf16* bb = Wt + (n0 + lm) * C_DIM + k + 16 * lg;
    v16bf b0 = *(const v16bf*)(bb);
    v16bf b1 = *(const v16bf*)(bb + 16 * C_DIM);
    v16bf b2 = *(const v16bf*)(bb + 32 * C_DIM);
    v16bf b3 = *(const v16bf*)(bb + 48 * C_DIM);
    c00 = WMMA_BF16(a0, b0, c00);  c01 = WMMA_BF16(a0, b1, c01);
    c02 = WMMA_BF16(a0, b2, c02);  c03 = WMMA_BF16(a0, b3, c03);
    c10 = WMMA_BF16(a1, b0, c10);  c11 = WMMA_BF16(a1, b1, c11);
    c12 = WMMA_BF16(a1, b2, c12);  c13 = WMMA_BF16(a1, b3, c13);
  }
  v8f cc[2][4] = {{c00, c01, c02, c03}, {c10, c11, c12, c13}};
#pragma unroll
  for (int i = 0; i < 2; ++i) {
#pragma unroll
    for (int dt = 0; dt < 4; ++dt) {
      const int n = n0 + dt * 16 + lm;
      const float bv = bias[n];
#pragma unroll
      for (int r = 0; r < 8; ++r) {
        const int m = m0 + 16 * i + r + 8 * lg;
        out[(size_t)m * C_DIM + n] = cc[i][dt][r] + bv;
      }
    }
  }
}

// ---------------------------------------------------------------- launch
extern "C" void kernel_launch(void* const* d_in, const int* in_sizes, int n_in,
                              void* d_out, int out_size, void* d_ws, size_t ws_size,
                              hipStream_t stream) {
  const float* hidden  = (const float*)d_in[0];   // [2,2048,1024]
  const float* W_attn  = (const float*)d_in[1];   // [1024,3072]
  const float* b_attn  = (const float*)d_in[2];   // [3072]
  const float* W_proj  = (const float*)d_in[3];   // [1024,1024]
  const float* b_proj  = (const float*)d_in[4];   // [1024]
  const float* kv_sc   = (const float*)d_in[5];
  const float* kv_zp   = (const float*)d_in[6];
  float* out = (float*)d_out;

  char* ws = (char*)d_ws;
  bf16* Xbf  = (bf16*)(ws);                                   //  8 MB
  bf16* Wt   = (bf16*)(ws + (size_t)8  * 1024 * 1024);        //  6 MB
  bf16* Wpt  = (bf16*)(ws + (size_t)14 * 1024 * 1024);        //  2 MB
  bf16* qbuf = (bf16*)(ws + (size_t)16 * 1024 * 1024);        //  8 MB
  bf16* kbuf = (bf16*)(ws + (size_t)24 * 1024 * 1024);        //  8 MB
  bf16* vtb  = (bf16*)(ws + (size_t)32 * 1024 * 1024);        //  8 MB
  bf16* att  = (bf16*)(ws + (size_t)40 * 1024 * 1024);        //  8 MB

  // 1. precision conversion + weight transposes
  {
    int n = M_TOK * C_DIM;
    cvt_f32_bf16<<<(n + 255) / 256, 256, 0, stream>>>(hidden, Xbf, n);
  }
  {
    int n = C_DIM * N_QKV;
    transpose_f32_bf16<<<(n + 255) / 256, 256, 0, stream>>>(W_attn, Wt, C_DIM, N_QKV);
  }
  {
    int n = C_DIM * C_DIM;
    transpose_f32_bf16<<<(n + 255) / 256, 256, 0, stream>>>(W_proj, Wpt, C_DIM, C_DIM);
  }

  // 2. QKV GEMM + fake-quant + layout for attention
  {
    int waves = (M_TOK / 32) * (N_QKV / 64);    // 128 * 48 = 6144
    qkv_wmma<<<waves / 8, 256, 0, stream>>>(Xbf, Wt, b_attn, kv_sc, kv_zp,
                                            qbuf, kbuf, vtb);
  }

  // 3. flash attention
  {
    int blocks = 2 * H_HEADS * (T_SEQ / 128);   // 32 * 16 = 512
    attn_wmma<<<blocks, 256, 0, stream>>>(qbuf, kbuf, vtb, att);
  }

  // 4. output projection
  {
    int waves = (M_TOK / 32) * (C_DIM / 64);    // 128 * 16 = 2048
    proj_wmma<<<waves / 8, 256, 0, stream>>>(att, Wpt, b_proj, out);
  }
}
